// XAttention_79448305041999
// MI455X (gfx1250) — matrix operand order repaired and checked by reading, not compiled, not run on hardware
//
#include <hip/hip_runtime.h>
#include <hip/hip_bf16.h>
#include <stdint.h>

#define DIMV   1024
#define HEADS  16
#define DH     64
#define BATCH  2
#define QLEN   1024
#define KLEN   4096
#define INNERV 4096

typedef __attribute__((ext_vector_type(16))) __bf16 bf16x16;
typedef __attribute__((ext_vector_type(8)))  float  f32x8;

union FragAB { uint32_t u[8]; bf16x16 v; };

__device__ __forceinline__ uint16_t f2bf(float f) {
  union { float f; uint32_t u; } x{f};
  uint32_t r = x.u + 0x7FFFu + ((x.u >> 16) & 1u);
  return (uint16_t)(r >> 16);
}
__device__ __forceinline__ uint32_t lds_lo32(const void* p) {
  return (uint32_t)(uintptr_t)p;   // flat->LDS: addr[31:0] is the LDS offset
}
__device__ __forceinline__ void async_b128(uint32_t lds, uint64_t gaddr) {
  asm volatile("global_load_async_to_lds_b128 %0, %1, off"
               :: "v"(lds), "v"(gaddr) : "memory");
}
__device__ __forceinline__ void wait_async0() {
  asm volatile("s_wait_asynccnt 0" ::: "memory");
}

// ---------------------------------------------------------------- converts
__global__ void k_cvt_bf16(const float* __restrict__ in, uint16_t* __restrict__ out, int n) {
  int i = blockIdx.x * blockDim.x + threadIdx.x;
  if (i < n) out[i] = f2bf(in[i]);
}

// weight convert + K-pair-interleave swizzle: out[((k>>1)*N + n)*2 + (k&1)] = in[k*N+n]
__global__ void k_cvt_swz_bf16(const float* __restrict__ in, uint16_t* __restrict__ out,
                               int nShift, int total) {
  int i = blockIdx.x * blockDim.x + threadIdx.x;
  if (i >= total) return;
  int k = i >> nShift;
  int n = i & ((1 << nShift) - 1);
  out[(((size_t)(k >> 1) << nShift) + n) * 2 + (k & 1)] = f2bf(in[i]);
}

// ---------------------------------------------------------------- layernorm (row = 1024)
__global__ __launch_bounds__(256) void k_layernorm_bf16(
    const float* __restrict__ x, const float* __restrict__ sc,
    const float* __restrict__ bi, uint16_t* __restrict__ out) {
  __shared__ float s_sum[256], s_sq[256];
  const int row = blockIdx.x;
  const float* xr = x + (size_t)row * DIMV;
  float v[4]; float s = 0.f, q = 0.f;
#pragma unroll
  for (int i = 0; i < 4; i++) { v[i] = xr[threadIdx.x + i*256]; s += v[i]; q += v[i]*v[i]; }
  s_sum[threadIdx.x] = s; s_sq[threadIdx.x] = q; __syncthreads();
  for (int off = 128; off > 0; off >>= 1) {
    if ((int)threadIdx.x < off) { s_sum[threadIdx.x] += s_sum[threadIdx.x+off]; s_sq[threadIdx.x] += s_sq[threadIdx.x+off]; }
    __syncthreads();
  }
  const float mu   = s_sum[0] * (1.0f/DIMV);
  const float var  = s_sq[0] * (1.0f/DIMV) - mu*mu;
  const float rstd = rsqrtf(var + 1e-6f);
#pragma unroll
  for (int i = 0; i < 4; i++) {
    int c = threadIdx.x + i*256;
    out[(size_t)row*DIMV + c] = f2bf((v[i]-mu)*rstd*sc[c] + bi[c]);
  }
}

// ---------------------------------------------------------------- kv + pos_embed[ids]
__global__ __launch_bounds__(256) void k_addpos_bf16(
    const float* __restrict__ kv, const float* __restrict__ pos,
    const int* __restrict__ ids, uint16_t* __restrict__ out) {
  const int row = blockIdx.x;
  const int id  = ids[row];
  const float* a = kv  + (size_t)row * DIMV;
  const float* p = pos + (size_t)id  * DIMV;
#pragma unroll
  for (int i = 0; i < 4; i++) {
    int c = threadIdx.x + i*256;
    out[(size_t)row*DIMV + c] = f2bf(a[c] + p[c]);
  }
}

// ---------------------------------------------------------------- WMMA GEMM
// C[M,N] = A[M,K](bf16 row-major) x Bsw[K,N](bf16, K-pair-interleaved).
// 64x128 block tile, 8 waves, double-buffered async global->LDS staging.
// EPI: 0 bf16 store | 1 fp32+residual | 2 gelu->bf16 | 3 keys/V-swizzle split (kv proj)
template<int EPI>
__global__ __launch_bounds__(256) void k_gemm_bf16_wmma(
    const uint16_t* __restrict__ A, const uint16_t* __restrict__ Bsw,
    uint16_t* __restrict__ Cbf, float* __restrict__ Cf,
    const float* __restrict__ Res, uint16_t* __restrict__ Cbf2,
    int M, int N, int K) {
  __shared__ alignas(16) uint32_t sA[2][64 * 16];    // [buf][row][kword]
  __shared__ alignas(16) uint32_t sB[2][16 * 128];   // [buf][kpair][n]
  const int tid = threadIdx.x, lane = tid & 31, wave = tid >> 5;
  const int half16 = (lane >> 4) & 1, l16 = lane & 15;
  const int wy = wave & 3, wx = wave >> 2;
  const int m0 = blockIdx.y * 64, n0 = blockIdx.x * 128;
  const uint32_t ldsA0 = lds_lo32(&sA[0][0]);
  const uint32_t ldsB0 = lds_lo32(&sB[0][0]);
  const uint64_t gA = (uint64_t)(uintptr_t)A;
  const uint64_t gB = (uint64_t)(uintptr_t)Bsw;

  auto issue_tile = [&](int k0, int buf) {
    {   // A tile 64x32 bf16 = 256 x 16B segments, one per thread
      int row = tid >> 2, s = tid & 3;
      async_b128(ldsA0 + buf*4096 + row*64 + s*16,
                 gA + ((size_t)(m0 + row)*K + k0)*2 + s*16);
    }
#pragma unroll
    for (int i = 0; i < 2; i++) {   // B tile: 16 kpair-rows x 512B = 512 segments
      int seg = tid + i*256; int kp = seg >> 5, s = seg & 31;
      async_b128(ldsB0 + buf*8192 + kp*512 + s*16,
                 gB + ((size_t)(k0/2 + kp)*N + n0)*4 + s*16);
    }
  };

  const f32x8 zero8 = {0.f,0.f,0.f,0.f,0.f,0.f,0.f,0.f};
  f32x8 acc[4] = {zero8, zero8, zero8, zero8};

  issue_tile(0, 0);
  int buf = 0;
  for (int k0 = 0; k0 < K; k0 += 32) {
    wait_async0();
    __syncthreads();
    if (k0 + 32 < K) issue_tile(k0 + 32, buf ^ 1);

    FragAB a;
#pragma unroll
    for (int v = 0; v < 8; v++) {
      int kk = (half16 ? 16 : 0) + v * 2;
      a.u[v] = sA[buf][(wy*16 + l16)*16 + (kk >> 1)];
    }
#pragma unroll
    for (int j = 0; j < 4; j++) {
      FragAB b;
      const int n = wx*64 + j*16 + l16;
#pragma unroll
      for (int v = 0; v < 8; v++) b.u[v] = sB[buf][(v + (half16 ? 8 : 0))*128 + n];
      acc[j] = __builtin_amdgcn_wmma_f32_16x16x32_bf16(
                 false, a.v, false, b.v, (short)0, acc[j], false, false);
    }
    buf ^= 1;
  }

#pragma unroll
  for (int j = 0; j < 4; j++) {
#pragma unroll
    for (int r = 0; r < 8; r++) {
      const int row = m0 + wy*16 + r + (half16 ? 8 : 0);
      const int col = n0 + wx*64 + j*16 + l16;
      const float vv = acc[j][r];
      if (EPI == 0) {
        Cbf[(size_t)row*N + col] = f2bf(vv);
      } else if (EPI == 1) {
        Cf[(size_t)row*N + col] = vv + Res[(size_t)row*N + col];
      } else if (EPI == 2) {
        const float x = vv;
        const float g = 0.5f * x * (1.0f + tanhf(0.7978845608f * (x + 0.044715f * x * x * x)));
        Cbf[(size_t)row*N + col] = f2bf(g);
      } else {   // EPI 3: kv projection; keys row-major, values key-pair-interleaved
        const int half = N >> 1;   // 1024
        if (col < half) {
          Cbf[(size_t)row*half + col] = f2bf(vv);
        } else {
          Cbf2[((size_t)(row >> 1)*half + (col - half))*2 + (row & 1)] = f2bf(vv);
        }
      }
    }
  }
}

// ---------------------------------------------------------------- flash attention
// grid(B*HEADS, QLEN/64), block 128 (4 waves); 16-query tile per wave, 32-key
// chunks; V chunks double-buffered via async DMA (pre-interleaved by key pair).
__global__ __launch_bounds__(128) void k_flash_attn(
    const uint16_t* __restrict__ Q,    // [B*QLEN, DIMV] bf16 projected queries
    const uint16_t* __restrict__ Kb,   // [B*KLEN, DIMV] bf16 keys, row-major
    const uint16_t* __restrict__ Vsw,  // [B*KLEN/2, DIMV, 2] bf16 values interleaved
    const unsigned char* __restrict__ mask,
    uint16_t* __restrict__ Ctx) {
  __shared__ alignas(16) uint32_t sV[2][16 * 64];
  __shared__ alignas(16) uint16_t sP[4 * 16 * 32];
  const int tid = threadIdx.x, lane = tid & 31, wave = tid >> 5;
  const int half16 = (lane >> 4) & 1, l16 = lane & 15;
  const int b = blockIdx.x >> 4, h = blockIdx.x & 15;
  const int q0 = blockIdx.y * 64 + wave * 16;
  const f32x8 zero8 = {0.f,0.f,0.f,0.f,0.f,0.f,0.f,0.f};
  const uint32_t ldsV0 = lds_lo32(&sV[0][0]);
  const uint64_t gV = (uint64_t)(uintptr_t)Vsw;

  auto issue_v = [&](int kc, int bufv) {
#pragma unroll
    for (int i = 0; i < 2; i++) {   // 16 kpair-rows x 256B (head slice) = 256 segs
      int seg = tid + i*128; int kp = seg >> 4, s = seg & 15;
      async_b128(ldsV0 + bufv*4096 + kp*256 + s*16,
                 gV + ((size_t)((b*KLEN + kc) >> 1) + kp)*(DIMV*4) + h*DH*4 + s*16);
    }
  };

  FragAB a_lo, a_hi;
  {
    const size_t qrow = (size_t)(b*QLEN + q0 + l16) * DIMV + h*DH;
#pragma unroll
    for (int v = 0; v < 8; v++) {
      int kk = (half16 ? 16 : 0) + v * 2;
      a_lo.u[v] = *(const uint32_t*)(Q + qrow + kk);
      a_hi.u[v] = *(const uint32_t*)(Q + qrow + 32 + kk);
    }
  }

  f32x8 cacc[4] = {zero8, zero8, zero8, zero8};
  float mrow[8], lrow[8];
#pragma unroll
  for (int r = 0; r < 8; r++) { mrow[r] = -1e30f; lrow[r] = 0.f; }

  issue_v(0, 0);
  int bufv = 0;
  for (int kc = 0; kc < KLEN; kc += 32) {
    wait_async0();
    __syncthreads();
    if (kc + 32 < KLEN) issue_v(kc + 32, bufv ^ 1);

    // S = Q Kt for 2 groups of 16 keys; row-major keys are already B-layout.
    f32x8 s0 = zero8, s1 = zero8;
    {
      FragAB bk;
      size_t krow = (size_t)(b*KLEN + kc + l16) * DIMV + h*DH;
#pragma unroll
      for (int v = 0; v < 8; v++) bk.u[v] = *(const uint32_t*)(Kb + krow + 2*v + 16*half16);
      s0 = __builtin_amdgcn_wmma_f32_16x16x32_bf16(false, a_lo.v, false, bk.v, (short)0, s0, false, false);
#pragma unroll
      for (int v = 0; v < 8; v++) bk.u[v] = *(const uint32_t*)(Kb + krow + 32 + 2*v + 16*half16);
      s0 = __builtin_amdgcn_wmma_f32_16x16x32_bf16(false, a_hi.v, false, bk.v, (short)0, s0, false, false);
      krow += (size_t)16 * DIMV;
#pragma unroll
      for (int v = 0; v < 8; v++) bk.u[v] = *(const uint32_t*)(Kb + krow + 2*v + 16*half16);
      s1 = __builtin_amdgcn_wmma_f32_16x16x32_bf16(false, a_lo.v, false, bk.v, (short)0, s1, false, false);
#pragma unroll
      for (int v = 0; v < 8; v++) bk.u[v] = *(const uint32_t*)(Kb + krow + 32 + 2*v + 16*half16);
      s1 = __builtin_amdgcn_wmma_f32_16x16x32_bf16(false, a_hi.v, false, bk.v, (short)0, s1, false, false);
    }

    const float mk0 = mask[b*KLEN + kc + l16]      ? 0.f : -1e30f;
    const float mk1 = mask[b*KLEN + kc + 16 + l16] ? 0.f : -1e30f;

#pragma unroll
    for (int r = 0; r < 8; r++) {
      float x0 = s0[r] * 0.125f + mk0;   // 1/sqrt(DH)
      float x1 = s1[r] * 0.125f + mk1;
      float mx = fmaxf(x0, x1);
#pragma unroll
      for (int o = 8; o >= 1; o >>= 1) mx = fmaxf(mx, __shfl_xor(mx, o, 32));
      const float mnew = fmaxf(mrow[r], mx);
      const float esc  = __expf(mrow[r] - mnew);
      const float p0   = __expf(x0 - mnew);
      const float p1   = __expf(x1 - mnew);
      float sr = p0 + p1;
#pragma unroll
      for (int o = 8; o >= 1; o >>= 1) sr += __shfl_xor(sr, o, 32);
      lrow[r] = lrow[r] * esc + sr;
      mrow[r] = mnew;
#pragma unroll
      for (int j = 0; j < 4; j++) cacc[j][r] *= esc;
      const int Mr = r + (half16 ? 8 : 0);
      sP[(wave*16 + Mr)*32 + l16]      = f2bf(p0);
      sP[(wave*16 + Mr)*32 + 16 + l16] = f2bf(p1);
    }

    FragAB pa;
#pragma unroll
    for (int v = 0; v < 8; v++) {
      int kk = (half16 ? 16 : 0) + v * 2;
      pa.u[v] = *(const uint32_t*)&sP[(wave*16 + l16)*32 + kk];
    }
#pragma unroll
    for (int j = 0; j < 4; j++) {
      FragAB vb;
#pragma unroll
      for (int v = 0; v < 8; v++) vb.u[v] = sV[bufv][(v + (half16 ? 8 : 0))*64 + j*16 + l16];
      cacc[j] = __builtin_amdgcn_wmma_f32_16x16x32_bf16(
                  false, pa.v, false, vb.v, (short)0, cacc[j], false, false);
    }
    bufv ^= 1;
  }

#pragma unroll
  for (int j = 0; j < 4; j++) {
#pragma unroll
    for (int r = 0; r < 8; r++) {
      const int row = b*QLEN + q0 + r + (half16 ? 8 : 0);
      const int col = h*DH + j*16 + l16;
      Ctx[(size_t)row*DIMV + col] = f2bf(cacc[j][r] / lrow[r]);
    }
  }
}

// ---------------------------------------------------------------- launch
extern "C" void kernel_launch(void* const* d_in, const int* in_sizes, int n_in,
                              void* d_out, int out_size, void* d_ws, size_t ws_size,
                              hipStream_t stream) {
  const float* q    = (const float*)d_in[0];
  const float* kv   = (const float*)d_in[1];
  const unsigned char* mask = (const unsigned char*)d_in[2];
  const int*   ids  = (const int*)d_in[3];
  const float* ln1s = (const float*)d_in[4];
  const float* ln1b = (const float*)d_in[5];
  const float* Wq   = (const float*)d_in[6];
  const float* Wkv  = (const float*)d_in[7];
  const float* Wo   = (const float*)d_in[8];
  const float* ln2s = (const float*)d_in[9];
  const float* ln2b = (const float*)d_in[10];
  const float* W1   = (const float*)d_in[11];
  const float* W2   = (const float*)d_in[12];
  const float* pos  = (const float*)d_in[13];
  float* out = (float*)d_out;

  char* ws = (char*)d_ws;
  size_t off = 0;
  auto alloc = [&](size_t bytes) { void* p = ws + off; off += (bytes + 255) & ~(size_t)255; return p; };

  uint16_t* Wq_b   = (uint16_t*)alloc((size_t)DIMV * DIMV * 2);
  uint16_t* Wkv_b  = (uint16_t*)alloc((size_t)DIMV * 2 * DIMV * 2);
  uint16_t* Wo_b   = (uint16_t*)alloc((size_t)DIMV * DIMV * 2);
  uint16_t* W1_b   = (uint16_t*)alloc((size_t)DIMV * INNERV * 2);
  uint16_t* W2_b   = (uint16_t*)alloc((size_t)INNERV * DIMV * 2);
  uint16_t* xln1   = (uint16_t*)alloc((size_t)BATCH * QLEN * DIMV * 2);
  uint16_t* qproj  = (uint16_t*)alloc((size_t)BATCH * QLEN * DIMV * 2);
  uint16_t* kvp    = (uint16_t*)alloc((size_t)BATCH * KLEN * DIMV * 2);
  uint16_t* keysb  = (uint16_t*)alloc((size_t)BATCH * KLEN * DIMV * 2);
  uint16_t* valsw  = (uint16_t*)alloc((size_t)BATCH * KLEN * DIMV * 2);
  uint16_t* ctx    = (uint16_t*)alloc((size_t)BATCH * QLEN * DIMV * 2);
  float*    attno  = (float*)   alloc((size_t)BATCH * QLEN * DIMV * 4);
  uint16_t* h2ln   = (uint16_t*)alloc((size_t)BATCH * QLEN * DIMV * 2);
  uint16_t* ffn1   = (uint16_t*)alloc((size_t)BATCH * QLEN * INNERV * 2);

  const int M = BATCH * QLEN;        // 2048
  const int Mkv = BATCH * KLEN;      // 8192

  // weights -> bf16, K-pair-interleaved for async B staging
  k_cvt_swz_bf16<<<(DIMV*DIMV)/256, 256, 0, stream>>>(Wq, Wq_b, 10, DIMV*DIMV);
  k_cvt_swz_bf16<<<(DIMV*2*DIMV)/256, 256, 0, stream>>>(Wkv, Wkv_b, 11, DIMV*2*DIMV);
  k_cvt_swz_bf16<<<(DIMV*DIMV)/256, 256, 0, stream>>>(Wo, Wo_b, 10, DIMV*DIMV);
  k_cvt_swz_bf16<<<(DIMV*INNERV)/256, 256, 0, stream>>>(W1, W1_b, 12, DIMV*INNERV);
  k_cvt_swz_bf16<<<(INNERV*DIMV)/256, 256, 0, stream>>>(W2, W2_b, 10, INNERV*DIMV);

  // LN1(q), kv + pos_embed[ids]
  k_layernorm_bf16<<<M, 256, 0, stream>>>(q, ln1s, ln1b, xln1);
  k_addpos_bf16<<<Mkv, 256, 0, stream>>>(kv, pos, ids, kvp);

  // projections
  k_gemm_bf16_wmma<0><<<dim3(DIMV/128, M/64), 256, 0, stream>>>(
      xln1, Wq_b, qproj, nullptr, nullptr, nullptr, M, DIMV, DIMV);
  k_gemm_bf16_wmma<3><<<dim3((2*DIMV)/128, Mkv/64), 256, 0, stream>>>(
      kvp, Wkv_b, keysb, nullptr, nullptr, valsw, Mkv, 2*DIMV, DIMV);

  // attention
  k_flash_attn<<<dim3(BATCH*HEADS, QLEN/64), 128, 0, stream>>>(qproj, keysb, valsw, mask, ctx);

  // output projection + residual (fp32)
  k_gemm_bf16_wmma<1><<<dim3(DIMV/128, M/64), 256, 0, stream>>>(
      ctx, Wo_b, nullptr, attno, q, nullptr, M, DIMV, DIMV);

  // FFN
  k_layernorm_bf16<<<M, 256, 0, stream>>>(attno, ln2s, ln2b, h2ln);
  k_gemm_bf16_wmma<2><<<dim3(INNERV/128, M/64), 256, 0, stream>>>(
      h2ln, W1_b, ffn1, nullptr, nullptr, nullptr, M, INNERV, DIMV);
  k_gemm_bf16_wmma<1><<<dim3(DIMV/128, M/64), 256, 0, stream>>>(
      ffn1, W2_b, nullptr, out, attno, nullptr, M, DIMV, INNERV);
}
